// EnhancedDFSMN_34127810134562
// MI455X (gfx1250) — compile-verified, matrix operand order checked by reading, not run on hardware
//
#include <hip/hip_runtime.h>
#include <cmath>

// ---------------------------------------------------------------------------
// Types for CDNA5 WMMA (gfx1250, wave32)
// ---------------------------------------------------------------------------
typedef __attribute__((ext_vector_type(16))) __bf16 v16bf;
typedef __attribute__((ext_vector_type(8)))  __bf16 bf16x8;
typedef __attribute__((ext_vector_type(8)))  float  f32x8;

#define IN_DIM 257
#define IN_PAD 288      // 257 padded to multiple of 32
#define T_LEN  1500
#define MEMT   50
#define LAT    5

// ---------------------------------------------------------------------------
// CDNA5 async global->LDS copy (ASYNCcnt-tracked DMA, no VGPR round trip).
// vdst = 32-bit LDS byte offset; vaddr = 64-bit global address.
// ---------------------------------------------------------------------------
__device__ __forceinline__ void async_ld_b128(unsigned lds_off, const void* gptr) {
    asm volatile("global_load_async_to_lds_b128 %0, %1, off"
                 :: "v"(lds_off), "v"(gptr) : "memory");
}
__device__ __forceinline__ void wait_async0() {
    asm volatile("s_wait_asynccnt 0x0" ::: "memory");
}
// Low 32 bits of the flat address of a __shared__ object == LDS byte offset
// (ISA 10.2 aperture rule).
__device__ __forceinline__ unsigned lds_off_of(const void* p) {
    return (unsigned)(size_t)p;
}

// ---------------------------------------------------------------------------
// f32[rows][K] -> bf16[rows][Kpad], zero padded columns
// ---------------------------------------------------------------------------
__global__ __launch_bounds__(256)
void k_cvt_pad(const float* __restrict__ in, __bf16* __restrict__ out,
               int K, int Kpad, size_t rows) {
    size_t i = (size_t)blockIdx.x * 256 + threadIdx.x;
    size_t total = rows * (size_t)Kpad;
    if (i < total) {
        int k   = (int)(i % (size_t)Kpad);
        size_t r = i / (size_t)Kpad;
        out[i] = (k < K) ? (__bf16)in[r * (size_t)K + k] : (__bf16)0.0f;
    }
}

// ---------------------------------------------------------------------------
// Weight transpose-convert: W[K][N] f32 -> Wt[Npad][Kpad] bf16 (K contiguous)
// ---------------------------------------------------------------------------
__global__ __launch_bounds__(256)
void k_cvt_wT(const float* __restrict__ W, __bf16* __restrict__ out,
              int K, int N, int Kpad, int Npad) {
    size_t i = (size_t)blockIdx.x * 256 + threadIdx.x;
    size_t total = (size_t)Npad * Kpad;
    if (i < total) {
        int k = (int)(i % (size_t)Kpad);
        int n = (int)(i / (size_t)Kpad);
        out[i] = (k < K && n < N) ? (__bf16)W[(size_t)k * N + n] : (__bf16)0.0f;
    }
}

// ---------------------------------------------------------------------------
// WMMA GEMM: C[M,N] = act(A[M,K] @ W[K,N] + bias)
//   A:  bf16, row stride lda (mult of 8, zero-padded to >= ceil32(K))
//   Wt: bf16, transposed [Npad][Kpad] (K contiguous, zero padded)
//   M multiple of 128. Block = 256 thr = 8 waves. Block tile 128(M) x 64(N);
//   each wave computes 32x32 (4 accumulators, A/B fragments reused).
//   Staging: double-buffered LDS filled by global_load_async_to_lds_b128;
//   one s_wait_asynccnt + one barrier per K-step, DMA overlaps WMMA.
//   act: 0=none 1=tanh 2=relu 3=sigmoid
// ---------------------------------------------------------------------------
__global__ __launch_bounds__(256)
void k_wmma_gemm(const __bf16* __restrict__ A, int lda,
                 const __bf16* __restrict__ Wt, int ldw,
                 const float* __restrict__ bias,
                 float* __restrict__ outF, __bf16* __restrict__ outB,
                 int M, int N, int K, int act) {
    __shared__ alignas(16) __bf16 As[2][128][40];   // 32 K + 8 pad (80B rows)
    __shared__ alignas(16) __bf16 Bs[2][64][40];    // transposed: [n][k]

    const int tid   = threadIdx.x;
    const int mBase = blockIdx.x * 128;
    const int nBase = blockIdx.y * 64;

    const int lane = tid & 31;
    const int wave = tid >> 5;
    const int wm32 = (wave & 3) << 5;   // wave M offset in tile (0..96)
    const int wn32 = (wave >> 2) << 5;  // wave N offset in tile (0 or 32)
    const int half = lane >> 4;
    const int l16  = lane & 15;

    // staging: per thread, 2 A chunks (rows r, r+64) + 1 B chunk of 16B
    const int srow = tid >> 2;          // 0..63
    const int sk   = (tid & 3) << 3;    // 0,8,16,24

    const __bf16* Ap0 = A  + (size_t)(mBase + srow) * lda + sk;
    const __bf16* Ap1 = Ap0 + (size_t)64 * lda;
    const __bf16* Bp  = Wt + (size_t)(nBase + srow) * ldw + sk;

    const unsigned ldsA0[2] = { lds_off_of(&As[0][srow][sk]),
                                lds_off_of(&As[1][srow][sk]) };
    const unsigned ldsA1[2] = { lds_off_of(&As[0][srow + 64][sk]),
                                lds_off_of(&As[1][srow + 64][sk]) };
    const unsigned ldsB[2]  = { lds_off_of(&Bs[0][srow][sk]),
                                lds_off_of(&Bs[1][srow][sk]) };

    f32x8 acc00 = {}, acc01 = {}, acc10 = {}, acc11 = {};

    const int kIters = (K + 31) >> 5;

    // prologue: DMA first tile into buffer 0
    async_ld_b128(ldsA0[0], Ap0);
    async_ld_b128(ldsA1[0], Ap1);
    async_ld_b128(ldsB[0],  Bp);

    for (int it = 0; it < kIters; ++it) {
        const int buf = it & 1;
        wait_async0();
        __syncthreads();                         // buf is ready for all waves

        if (it + 1 < kIters) {                   // DMA next tile into other buf
            const int k1 = (it + 1) << 5;
            const int nb = buf ^ 1;
            async_ld_b128(ldsA0[nb], Ap0 + k1);
            async_ld_b128(ldsA1[nb], Ap1 + k1);
            async_ld_b128(ldsB[nb],  Bp  + k1);
        }

        // fragments: lane holds K(e)=16*(e>>3)+8*half+(e&7)
        const __bf16 (*Ab)[40] = As[buf];
        const __bf16 (*Bb)[40] = Bs[buf];
        union { v16bf v; bf16x8 h[2]; } a0, a1, b0, b1;
        a0.h[0] = *(const bf16x8*)&Ab[wm32 + l16][(half << 3)];
        a0.h[1] = *(const bf16x8*)&Ab[wm32 + l16][16 + (half << 3)];
        a1.h[0] = *(const bf16x8*)&Ab[wm32 + 16 + l16][(half << 3)];
        a1.h[1] = *(const bf16x8*)&Ab[wm32 + 16 + l16][16 + (half << 3)];
        b0.h[0] = *(const bf16x8*)&Bb[wn32 + l16][(half << 3)];
        b0.h[1] = *(const bf16x8*)&Bb[wn32 + l16][16 + (half << 3)];
        b1.h[0] = *(const bf16x8*)&Bb[wn32 + 16 + l16][(half << 3)];
        b1.h[1] = *(const bf16x8*)&Bb[wn32 + 16 + l16][16 + (half << 3)];

        acc00 = __builtin_amdgcn_wmma_f32_16x16x32_bf16(
            false, a0.v, false, b0.v, (short)0, acc00, false, false);
        acc01 = __builtin_amdgcn_wmma_f32_16x16x32_bf16(
            false, a0.v, false, b1.v, (short)0, acc01, false, false);
        acc10 = __builtin_amdgcn_wmma_f32_16x16x32_bf16(
            false, a1.v, false, b0.v, (short)0, acc10, false, false);
        acc11 = __builtin_amdgcn_wmma_f32_16x16x32_bf16(
            false, a1.v, false, b1.v, (short)0, acc11, false, false);
        // wave's ds reads complete before its next barrier signal (DScnt wait
        // forced by the WMMA consumers), so one barrier per iteration is safe.
    }

    // ---- epilogue: VGPR r holds M = r + 8*half ----
    f32x8 accs[2][2] = { {acc00, acc01}, {acc10, acc11} };
#pragma unroll
    for (int ms = 0; ms < 2; ++ms) {
#pragma unroll
        for (int ns = 0; ns < 2; ++ns) {
            const int n = nBase + wn32 + (ns << 4) + l16;
            if (n >= N) continue;
            const float bv = bias ? bias[n] : 0.0f;
#pragma unroll
            for (int r = 0; r < 8; ++r) {
                int m = mBase + wm32 + (ms << 4) + (half << 3) + r;
                float v = accs[ms][ns][r] + bv;
                if      (act == 1) v = tanhf(v);
                else if (act == 2) v = fmaxf(v, 0.0f);
                else if (act == 3) v = 1.0f / (1.0f + expf(-v));
                size_t o = (size_t)m * N + n;
                if (outF) outF[o] = v;
                if (outB) outB[o] = (__bf16)v;
            }
        }
    }
}

// ---------------------------------------------------------------------------
// Row softmax over N (=257), in place. One block per row.
// ---------------------------------------------------------------------------
__global__ __launch_bounds__(256)
void k_softmax_row(float* __restrict__ a, int N) {
    float* p = a + (size_t)blockIdx.x * N;
    const int tid = threadIdx.x;
    __shared__ float red[256];

    float v1 = (tid < N)       ? p[tid]       : -3.0e38f;
    float v2 = (tid + 256 < N) ? p[tid + 256] : -3.0e38f;

    red[tid] = fmaxf(v1, v2);
    __syncthreads();
    for (int o = 128; o > 0; o >>= 1) {
        if (tid < o) red[tid] = fmaxf(red[tid], red[tid + o]);
        __syncthreads();
    }
    float m = red[0];
    __syncthreads();

    float e1 = (tid < N)       ? expf(v1 - m) : 0.0f;
    float e2 = (tid + 256 < N) ? expf(v2 - m) : 0.0f;
    red[tid] = e1 + e2;
    __syncthreads();
    for (int o = 128; o > 0; o >>= 1) {
        if (tid < o) red[tid] += red[tid + o];
        __syncthreads();
    }
    float inv = 1.0f / red[0];
    if (tid < N)       p[tid]       = e1 * inv;
    if (tid + 256 < N) p[tid + 256] = e2 * inv;
}

// ---------------------------------------------------------------------------
// ff = x * attn * gate : f32 out stride 257, bf16 out stride 288 (zero pad)
// ---------------------------------------------------------------------------
__global__ __launch_bounds__(256)
void k_ff_mul(const float* __restrict__ x, const float* __restrict__ attn,
              const float* __restrict__ gate, float* __restrict__ ff,
              __bf16* __restrict__ ffb, size_t nPadTotal) {
    size_t i = (size_t)blockIdx.x * 256 + threadIdx.x;
    if (i < nPadTotal) {
        size_t r = i / IN_PAD;
        int    c = (int)(i % IN_PAD);
        if (c < IN_DIM) {
            size_t j = r * IN_DIM + c;
            float v = x[j] * attn[j] * gate[j];
            ff[j]  = v;
            ffb[i] = (__bf16)v;
        } else {
            ffb[i] = (__bf16)0.0f;
        }
    }
}

// ---------------------------------------------------------------------------
// Notch filters: out = ff * prod_i resp_i -> bf16 feats (stride 288, padded)
// ---------------------------------------------------------------------------
__global__ __launch_bounds__(256)
void k_notch(const float* __restrict__ ff, const float* __restrict__ npar,
             __bf16* __restrict__ out) {
    const size_t tok = blockIdx.x;
    const float* np = npar + tok * 8;
    float c[4], b2[4];
#pragma unroll
    for (int i = 0; i < 4; ++i) {
        c[i] = np[i];
        float r  = np[4 + i];
        float sp = (r > 15.0f) ? r : log1pf(expf(r));
        float bw = sp * 1.3f;
        b2[i] = 2.0f * bw * bw;
    }
    for (int f = threadIdx.x; f < IN_PAD; f += 256) {
        float v = 0.0f;
        if (f < IN_DIM) {
            float fi = (float)f * (1.0f / 256.0f);
            v = ff[tok * IN_DIM + f];
            if (fi > 0.1f) {
#pragma unroll
                for (int i = 0; i < 4; ++i) {
                    float d = fi - c[i];
                    v *= (1.0f - expf(-(d * d) / b2[i]));
                }
            }
        }
        out[tok * IN_PAD + f] = (__bf16)v;
    }
}

// ---------------------------------------------------------------------------
// Per-layer tap sums: taps[0..49]=sum_d mem_w, taps[50..54]=sum_d la_w
// ---------------------------------------------------------------------------
__global__ void k_tap_sums(const float* __restrict__ memw,
                           const float* __restrict__ law,
                           float* __restrict__ taps, int D) {
    int t = threadIdx.x;
    if (t < MEMT) {
        float s = 0.0f;
        for (int d = 0; d < D; ++d) s += memw[(size_t)t * D + d];
        taps[t] = s;
    } else if (t < MEMT + LAT) {
        int u = t - MEMT;
        float s = 0.0f;
        for (int d = 0; d < D; ++d) s += law[(size_t)u * D + d];
        taps[t] = s;
    }
}

// ---------------------------------------------------------------------------
// DFSMN context + LayerNorm + skip + relu. One block per token.
// Early region (t<50) uses absolute-time tap indexing per reference.
// ---------------------------------------------------------------------------
__global__ __launch_bounds__(256)
void k_dfsmn_ctx_ln(const float* __restrict__ h, const float* __restrict__ skip,
                    const float* __restrict__ taps, const float* __restrict__ g,
                    const float* __restrict__ be, __bf16* __restrict__ outbf,
                    int T, int D) {
    __shared__ float sh[512];
    __shared__ float red[256];
    __shared__ float stats[2];

    const int tok = blockIdx.x;
    const int b = tok / T;
    const int t = tok - b * T;
    const size_t base = (size_t)b * T * D;
    const float* hrow = h + base + (size_t)t * D;

    for (int d = threadIdx.x; d < D; d += 256) {
        float acc = hrow[d];
        if (t < MEMT) {
            for (int s = 0; s < t; ++s)
                acc += taps[s] * h[base + (size_t)s * D + d];
        } else {
            const float* hp = h + base + (size_t)(t - MEMT) * D + d;
#pragma unroll 5
            for (int tau = 0; tau < MEMT; ++tau)
                acc += taps[tau] * hp[(size_t)tau * D];
        }
        int lmax = T - 1 - t; if (lmax > LAT) lmax = LAT;
        const float* hf = hrow + D + d;
        for (int tau = 0; tau < lmax; ++tau)
            acc += taps[MEMT + tau] * hf[(size_t)tau * D];
        sh[d] = acc;
    }
    __syncthreads();

    float s = 0.0f, q = 0.0f;
    for (int d = threadIdx.x; d < D; d += 256) {
        float v = sh[d]; s += v; q += v * v;
    }
    red[threadIdx.x] = s;
    __syncthreads();
    for (int o = 128; o > 0; o >>= 1) {
        if (threadIdx.x < o) red[threadIdx.x] += red[threadIdx.x + o];
        __syncthreads();
    }
    if (threadIdx.x == 0) stats[0] = red[0];
    __syncthreads();
    red[threadIdx.x] = q;
    __syncthreads();
    for (int o = 128; o > 0; o >>= 1) {
        if (threadIdx.x < o) red[threadIdx.x] += red[threadIdx.x + o];
        __syncthreads();
    }
    if (threadIdx.x == 0) stats[1] = red[0];
    __syncthreads();

    const float mean = stats[0] / (float)D;
    const float var  = stats[1] / (float)D - mean * mean;
    const float inv  = rsqrtf(var + 1e-5f);
    const size_t ob  = (size_t)tok * D;

    for (int d = threadIdx.x; d < D; d += 256) {
        float v = (sh[d] - mean) * inv * g[d] + be[d] + skip[ob + d];
        v = fmaxf(v, 0.0f);
        outbf[ob + d] = (__bf16)v;
    }
}

// ---------------------------------------------------------------------------
// Final: bm = bp>0.6 ; out = (o*(1-bm) + x*bm) * og
// ---------------------------------------------------------------------------
__global__ __launch_bounds__(256)
void k_final_mix(const float* __restrict__ x, const float* __restrict__ o,
                 const float* __restrict__ og, const float* __restrict__ bp,
                 float* __restrict__ out, size_t n) {
    size_t i = (size_t)blockIdx.x * 256 + threadIdx.x;
    if (i < n) {
        float bm = (bp[i] > 0.6f) ? 1.0f : 0.0f;
        float oo = o[i] * (1.0f - bm) + x[i] * bm;
        out[i] = oo * og[i];
    }
}

// ---------------------------------------------------------------------------
// Host helpers
// ---------------------------------------------------------------------------
static inline int ceil_to(int x, int a) { return (x + a - 1) / a * a; }

static void run_gemm(hipStream_t stream, const __bf16* A, int lda,
                     const float* W, const float* bias,
                     float* outF, __bf16* outB, __bf16* wbf,
                     int M, int N, int K, int act) {
    const int Kpad = ceil_to(K, 32);
    const int Npad = ceil_to(N, 64);
    size_t nw = (size_t)Npad * Kpad;
    k_cvt_wT<<<dim3((unsigned)((nw + 255) / 256)), dim3(256), 0, stream>>>(
        W, wbf, K, N, Kpad, Npad);
    dim3 grid((unsigned)(M / 128), (unsigned)(Npad / 64));
    k_wmma_gemm<<<grid, dim3(256), 0, stream>>>(A, lda, wbf, Kpad, bias,
                                                outF, outB, M, N, K, act);
}

// ---------------------------------------------------------------------------
// kernel_launch
// ---------------------------------------------------------------------------
extern "C" void kernel_launch(void* const* d_in, const int* in_sizes, int n_in,
                              void* d_out, int out_size, void* d_ws, size_t ws_size,
                              hipStream_t stream) {
    if (n_in < 53) return;

    const float* X = (const float*)d_in[0];
    const size_t NT = (size_t)in_sizes[0] / IN_DIM;     // 48000, multiple of 128
    const int M = (int)NT;
    const int T = T_LEN;

    const float* fa_w1 = (const float*)d_in[1];  const float* fa_b1 = (const float*)d_in[2];
    const float* fa_w2 = (const float*)d_in[3];  const float* fa_b2 = (const float*)d_in[4];
    const float* fg_w1 = (const float*)d_in[5];  const float* fg_b1 = (const float*)d_in[6];
    const float* fg_w2 = (const float*)d_in[7];  const float* fg_b2 = (const float*)d_in[8];
    const float* nf_w1 = (const float*)d_in[9];  const float* nf_b1 = (const float*)d_in[10];
    const float* nf_w2 = (const float*)d_in[11]; const float* nf_b2 = (const float*)d_in[12];
    const float* bd_w1 = (const float*)d_in[13]; const float* bd_b1 = (const float*)d_in[14];
    const float* bd_w2 = (const float*)d_in[15]; const float* bd_b2 = (const float*)d_in[16];
    const float* out_w  = (const float*)d_in[49]; const float* out_b  = (const float*)d_in[50];
    const float* gate_w = (const float*)d_in[51]; const float* gate_b = (const float*)d_in[52];

    const int HID[5] = {IN_DIM, 512, 384, 256, 128};

    // workspace carve-out
    char* ws = (char*)d_ws;
    size_t ofs = 0;
    auto carve = [&](size_t bytes) -> char* {
        char* p = ws + ofs;
        ofs += (bytes + 255) & ~(size_t)255;
        return p;
    };
    float*  bufA  = (float*)carve(NT * 512 * sizeof(float));
    float*  bufB  = (float*)carve(NT * 512 * sizeof(float));
    float*  bufC  = (float*)carve(NT * 512 * sizeof(float));
    float*  ffF   = (float*)carve(NT * IN_DIM * sizeof(float));
    float*  nparF = (float*)carve(NT * 8 * sizeof(float));
    __bf16* xbf   = (__bf16*)carve(NT * IN_PAD * sizeof(__bf16));
    __bf16* bf1   = (__bf16*)carve(NT * 512 * sizeof(__bf16));
    __bf16* bf2   = (__bf16*)carve(NT * 512 * sizeof(__bf16));
    __bf16* wbf   = (__bf16*)carve(512 * 512 * sizeof(__bf16));
    float*  taps  = (float*)carve(4 * 64 * sizeof(float));
    (void)ws_size;

    float* d_out1 = (float*)d_out;                 // o * og
    float* d_bp   = d_out1 + NT * IN_DIM;          // bp

    const size_t nElem = NT * IN_DIM;
    const size_t nPad  = NT * IN_PAD;
    const unsigned nBlk    = (unsigned)((nElem + 255) / 256);
    const unsigned nBlkPad = (unsigned)((nPad + 255) / 256);

    // x -> bf16 (stride 288, zero padded)
    k_cvt_pad<<<dim3(nBlkPad), dim3(256), 0, stream>>>(X, xbf, IN_DIM, IN_PAD, NT);

    // freq attention: attn = softmax(tanh(x@fa_w1+b1)@fa_w2+b2)
    run_gemm(stream, xbf, IN_PAD, fa_w1, fa_b1, nullptr, bf1, wbf, M, 512, IN_DIM, 1);
    run_gemm(stream, bf1, 512,    fa_w2, fa_b2, bufA, nullptr, wbf, M, IN_DIM, 512, 0);
    k_softmax_row<<<dim3((unsigned)NT), dim3(256), 0, stream>>>(bufA, IN_DIM);

    // freq gate: gate = sigmoid(relu(x@fg_w1+b1)@fg_w2+b2)
    run_gemm(stream, xbf, IN_PAD, fg_w1, fg_b1, nullptr, bf1, wbf, M, 512, IN_DIM, 2);
    run_gemm(stream, bf1, 512,    fg_w2, fg_b2, bufB, nullptr, wbf, M, IN_DIM, 512, 3);

    // ff = x * attn * gate (f32 stride 257, bf16 stride 288 in bf1)
    k_ff_mul<<<dim3(nBlkPad), dim3(256), 0, stream>>>(X, bufA, bufB, ffF, bf1, nPad);

    // notch params: npar = relu(ff@nf_w1+b1)@nf_w2+b2
    run_gemm(stream, bf1, IN_PAD, nf_w1, nf_b1, nullptr, bf2, wbf, M, 128, IN_DIM, 2);
    run_gemm(stream, bf2, 128,    nf_w2, nf_b2, nparF, nullptr, wbf, M, 8, 128, 0);

    // bypass prob: bp = sigmoid(relu(x@bd_w1+b1)@bd_w2+b2) -> second output
    run_gemm(stream, xbf, IN_PAD, bd_w1, bd_b1, nullptr, bf1, wbf, M, 128, IN_DIM, 2);
    run_gemm(stream, bf1, 128,    bd_w2, bd_b2, d_bp, nullptr, wbf, M, IN_DIM, 128, 3);

    // notch filtering -> feats (bf16 stride 288) in bf2
    k_notch<<<dim3((unsigned)NT), dim3(256), 0, stream>>>(ffF, nparF, bf2);

    // DFSMN stack
    __bf16* cur = bf2;
    __bf16* nxt = bf1;
    int ldaCur = IN_PAD;
    for (int i = 0; i < 4; ++i) {
        const int din  = HID[i];
        const int dout = HID[i + 1];
        const int pbase = 17 + i * 8;
        const float* l_w   = (const float*)d_in[pbase + 0];
        const float* l_b   = (const float*)d_in[pbase + 1];
        const float* l_mem = (const float*)d_in[pbase + 2];
        const float* l_la  = (const float*)d_in[pbase + 3];
        const float* l_g   = (const float*)d_in[pbase + 4];
        const float* l_be  = (const float*)d_in[pbase + 5];
        const float* s_w   = (const float*)d_in[pbase + 6];
        const float* s_b   = (const float*)d_in[pbase + 7];

        k_tap_sums<<<dim3(1), dim3(64), 0, stream>>>(l_mem, l_la, taps + i * 64, dout);
        run_gemm(stream, xbf, IN_PAD, s_w, s_b, bufB, nullptr, wbf, M, dout, IN_DIM, 0);
        run_gemm(stream, cur, ldaCur, l_w, l_b, bufC, nullptr, wbf, M, dout, din, 0);
        k_dfsmn_ctx_ln<<<dim3((unsigned)NT), dim3(256), 0, stream>>>(
            bufC, bufB, taps + i * 64, l_g, l_be, nxt, T, dout);

        __bf16* tmp = cur; cur = nxt; nxt = tmp;
        ldaCur = dout;
    }

    // output heads: o = feats@out_w+b ; og = sigmoid(feats@gate_w+b)
    run_gemm(stream, cur, ldaCur, out_w,  out_b,  bufA, nullptr, wbf, M, IN_DIM, 128, 0);
    run_gemm(stream, cur, ldaCur, gate_w, gate_b, bufB, nullptr, wbf, M, IN_DIM, 128, 3);

    // final mix
    k_final_mix<<<dim3(nBlk), dim3(256), 0, stream>>>(X, bufA, bufB, d_bp, d_out1, nElem);
}
